// Net_16381005267357
// MI455X (gfx1250) — matrix operand order repaired and checked by reading, not compiled, not run on hardware
//
#include <hip/hip_runtime.h>
#include <hip/hip_bf16.h>

typedef __attribute__((ext_vector_type(16))) __bf16 v16bf;
typedef __attribute__((ext_vector_type(8)))  __bf16 v8bf;
typedef __attribute__((ext_vector_type(4)))  __bf16 v4bf;
typedef __attribute__((ext_vector_type(8)))  float  v8f;

static constexpr int N_  = 100000;
static constexpr int E_  = 1600000;
static constexpr int F_  = 128;
static constexpr int H_  = 128;
static constexpr int L_  = 4;
static constexpr int G_  = 512;
static constexpr int O_  = 10;
static constexpr int DIN_ = F_ + L_ * H_;   // 640
static constexpr int STR  = H_ + 8;         // padded LDS row stride in bf16

// ---------------------------------------------------------------------------
// Pack W1s/W2s (f32 [L][128][128], row-major W[k][n]) into WMMA B-fragment
// layout, bf16.  Fragment (kt,nt): 32 lanes x 16 contiguous bf16 per lane.
// Lane: n = lane&15 (column nt*16+n), half = lane>>4; element e -> K = kt*32
// + (e>>3)*16 + half*8 + (e&7)   (matches A fragment k order).
// ---------------------------------------------------------------------------
__global__ void pack_weights(const float* __restrict__ W1s,
                             const float* __restrict__ W2s,
                             __bf16* __restrict__ Wp) {
    int gid = blockIdx.x * blockDim.x + threadIdx.x;   // 2*L*32*32*16 = 131072
    if (gid >= 2 * L_ * 32 * 32 * 16) return;
    int e     = gid & 15;
    int lane  = (gid >> 4) & 31;
    int frag  = (gid >> 9) & 31;       // kt*8 + nt
    int layer = (gid >> 14) & 3;
    int which = gid >> 16;
    int kt = frag >> 3, nt = frag & 7;
    int col = nt * 16 + (lane & 15);
    int k   = kt * 32 + (e >> 3) * 16 + (lane >> 4) * 8 + (e & 7);
    const float* W = (which ? W2s : W1s) + layer * H_ * H_;
    Wp[gid] = (__bf16)W[k * H_ + col];
}

// ---------------------------------------------------------------------------
// pooled[0] = segment_sum(x, batch): one thread per (row, 4 columns)
// ---------------------------------------------------------------------------
__global__ void pool_x(const float* __restrict__ x, const int* __restrict__ batch,
                       float* __restrict__ emb) {
    int gid = blockIdx.x * blockDim.x + threadIdx.x;
    if (gid >= N_ * 32) return;
    int row = gid >> 5, c4 = (gid & 31) * 4;
    float4 v = *(const float4*)(x + row * F_ + c4);
    float* dst = emb + batch[row] * DIN_ + c4;
    unsafeAtomicAdd(dst + 0, v.x);
    unsafeAtomicAdd(dst + 1, v.y);
    unsafeAtomicAdd(dst + 2, v.z);
    unsafeAtomicAdd(dst + 3, v.w);
}

// ---------------------------------------------------------------------------
// Edge aggregation: one wave per edge, lane handles 4 floats (32*4 = 128).
// agg[dst] += h[src].  agg (51 MB) fits in the 192 MB L2, so the fp32
// atomics stay on-chip.
// ---------------------------------------------------------------------------
__global__ void edge_agg(const float* __restrict__ h, const int* __restrict__ ei,
                         float* __restrict__ agg) {
    int wid  = (blockIdx.x * blockDim.x + threadIdx.x) >> 5;
    int lane = threadIdx.x & 31;
    if (wid >= E_) return;
    int s = ei[wid];
    int d = ei[E_ + wid];
    float4 v = *(const float4*)(h + (size_t)s * H_ + lane * 4);
    float* dst = agg + (size_t)d * H_ + lane * 4;
    unsafeAtomicAdd(dst + 0, v.x);
    unsafeAtomicAdd(dst + 1, v.y);
    unsafeAtomicAdd(dst + 2, v.z);
    unsafeAtomicAdd(dst + 3, v.w);
}

// ---------------------------------------------------------------------------
// Fused GIN MLP: z = (1+eps)*h + agg; z1 = relu(z@W1+b1); z2 = z1@W2+b2.
// 256 threads = 8 waves; block handles 128 rows; wave w owns rows [w*16,w*16+16).
// Emits 64 v_wmma_f32_16x16x32_bf16 per wave. Also produces per-column
// sum / sum-of-squares partials for BatchNorm.  zout may alias agg.
// ---------------------------------------------------------------------------
__device__ inline v16bf load_a_frag(const __bf16* buf, int rb, int kt, int lane) {
    int m = lane & 15, half = lane >> 4;
    const __bf16* p0 = buf + (rb + m) * STR + kt * 32 + half * 8;
    v8bf lo = *(const v8bf*)p0;          // K = kt*32 + half*8 + 0..7
    v8bf hi = *(const v8bf*)(p0 + 16);   // K = kt*32 + 16 + half*8 + 0..7
    v16bf a;
#pragma unroll
    for (int i = 0; i < 8; ++i) { a[i] = lo[i]; a[i + 8] = hi[i]; }
    return a;
}

__launch_bounds__(256)
__global__ void gin_mlp(const float* __restrict__ hin, const float* __restrict__ agg,
                        const __bf16* __restrict__ Wp, int layer,
                        const float* __restrict__ b1s, const float* __restrict__ b2s,
                        const float* __restrict__ epsArr,
                        float* __restrict__ zout, float* __restrict__ gsum,
                        float* __restrict__ gsq) {
    __shared__ __bf16 zA[128 * STR];
    __shared__ __bf16 z1[128 * STR];
    __shared__ float  red[2][8][128];

    const int tid = threadIdx.x;
    const int lane = tid & 31, w = tid >> 5;
    const int nn = lane & 15, half = lane >> 4;
    const int rowBase = blockIdx.x * 128;
    const float epsv = 1.0f + epsArr[layer];
    const float* b1 = b1s + layer * H_;
    const float* b2 = b2s + layer * H_;

    // ---- Phase 1: build A-operand zA (bf16) in LDS ----
#pragma unroll
    for (int i = 0; i < 16; ++i) {
        int idx4 = i * 256 + tid;              // 4096 float4 groups
        int r = idx4 >> 5, c4 = (idx4 & 31) * 4;
        int grow = rowBase + r;
        float4 hv = make_float4(0.f, 0.f, 0.f, 0.f), av = hv;
        if (grow < N_) {
            hv = *(const float4*)(hin + (size_t)grow * H_ + c4);
            av = *(const float4*)(agg + (size_t)grow * H_ + c4);
        }
        v4bf pk;
        pk[0] = (__bf16)(epsv * hv.x + av.x);
        pk[1] = (__bf16)(epsv * hv.y + av.y);
        pk[2] = (__bf16)(epsv * hv.z + av.z);
        pk[3] = (__bf16)(epsv * hv.w + av.w);
        *(v4bf*)(&zA[r * STR + c4]) = pk;
    }
    __syncthreads();

    // ---- Phase 2: GEMM1 + bias + ReLU -> z1 (bf16, LDS) ----
    {
        v16bf a0 = load_a_frag(zA, w * 16, 0, lane);
        v16bf a1 = load_a_frag(zA, w * 16, 1, lane);
        v16bf a2 = load_a_frag(zA, w * 16, 2, lane);
        v16bf a3 = load_a_frag(zA, w * 16, 3, lane);
        const __bf16* W1p = Wp + (size_t)(layer * 32) * 512;
#pragma unroll
        for (int nt = 0; nt < 8; ++nt) {
            v8f acc = {};
#pragma unroll
            for (int kt = 0; kt < 4; ++kt) {
                v16bf b = *(const v16bf*)(W1p + (kt * 8 + nt) * 512 + lane * 16);
                v16bf a = (kt == 0) ? a0 : (kt == 1) ? a1 : (kt == 2) ? a2 : a3;
                acc = __builtin_amdgcn_wmma_f32_16x16x32_bf16(
                    false, a, false, b, (short)0, acc, false, false);
            }
            int col = nt * 16 + nn;
            float bias = b1[col];
#pragma unroll
            for (int r = 0; r < 8; ++r) {
                float v = acc[r] + bias;
                v = v > 0.f ? v : 0.f;
                z1[(w * 16 + r + half * 8) * STR + col] = (__bf16)v;
            }
        }
    }
    __syncthreads();

    // ---- Phase 3: GEMM2 + bias; store z2, accumulate BN partials ----
    {
        v16bf a0 = load_a_frag(z1, w * 16, 0, lane);
        v16bf a1 = load_a_frag(z1, w * 16, 1, lane);
        v16bf a2 = load_a_frag(z1, w * 16, 2, lane);
        v16bf a3 = load_a_frag(z1, w * 16, 3, lane);
        const __bf16* W2p = Wp + (size_t)((L_ + layer) * 32) * 512;
#pragma unroll
        for (int nt = 0; nt < 8; ++nt) {
            v8f acc = {};
#pragma unroll
            for (int kt = 0; kt < 4; ++kt) {
                v16bf b = *(const v16bf*)(W2p + (kt * 8 + nt) * 512 + lane * 16);
                v16bf a = (kt == 0) ? a0 : (kt == 1) ? a1 : (kt == 2) ? a2 : a3;
                acc = __builtin_amdgcn_wmma_f32_16x16x32_bf16(
                    false, a, false, b, (short)0, acc, false, false);
            }
            int col = nt * 16 + nn;
            float bias = b2[col];
            float s = 0.f, sq = 0.f;
#pragma unroll
            for (int r = 0; r < 8; ++r) {
                int grow = rowBase + w * 16 + r + half * 8;
                float v = acc[r] + bias;
                if (grow < N_) {
                    zout[(size_t)grow * H_ + col] = v;
                    s += v;
                    sq += v * v;
                }
            }
            s  += __shfl_xor(s, 16, 32);
            sq += __shfl_xor(sq, 16, 32);
            if (half == 0) { red[0][w][col] = s; red[1][w][col] = sq; }
        }
    }
    __syncthreads();
    if (tid < 128) {
        float s = 0.f, sq = 0.f;
#pragma unroll
        for (int w2 = 0; w2 < 8; ++w2) { s += red[0][w2][tid]; sq += red[1][w2][tid]; }
        unsafeAtomicAdd(&gsum[tid], s);
        unsafeAtomicAdd(&gsq[tid], sq);
    }
}

// ---------------------------------------------------------------------------
// BatchNorm + ReLU + per-graph pooling: h = relu(bn(z)); emb[batch] += h.
// ---------------------------------------------------------------------------
__global__ void bn_relu_pool(const float* __restrict__ z, const float* __restrict__ gsum,
                             const float* __restrict__ gsq,
                             const float* __restrict__ gammas, const float* __restrict__ betas,
                             int layer, const int* __restrict__ batch,
                             float* __restrict__ hout, float* __restrict__ emb) {
    int gid = blockIdx.x * blockDim.x + threadIdx.x;
    if (gid >= N_ * 32) return;
    int row = gid >> 5, c4 = (gid & 31) * 4;
    const float invN = 1.0f / (float)N_;
    float4 zv = *(const float4*)(z + (size_t)row * H_ + c4);
    float zz[4] = {zv.x, zv.y, zv.z, zv.w};
    float* e = emb + batch[row] * DIN_ + F_ + layer * H_ + c4;
    float hh[4];
#pragma unroll
    for (int c = 0; c < 4; ++c) {
        int col = c4 + c;
        float mu  = gsum[col] * invN;
        float var = gsq[col] * invN - mu * mu;
        float rstd = rsqrtf(var + 1e-5f);
        float v = (zz[c] - mu) * rstd * gammas[layer * H_ + col] + betas[layer * H_ + col];
        v = v > 0.f ? v : 0.f;
        hh[c] = v;
        unsafeAtomicAdd(e + c, v);
    }
    *(float4*)(hout + (size_t)row * H_ + c4) = make_float4(hh[0], hh[1], hh[2], hh[3]);
}

// ---------------------------------------------------------------------------
// Final classifier: out = emb @ Wout + bout   ([512,640] x [640,10])
// ---------------------------------------------------------------------------
__global__ void classify(const float* __restrict__ emb, const float* __restrict__ Wout,
                         const float* __restrict__ bout, float* __restrict__ out) {
    int gid = blockIdx.x * blockDim.x + threadIdx.x;
    if (gid >= G_ * O_) return;
    int g = gid / O_, o = gid % O_;
    float s = bout[o];
    for (int d = 0; d < DIN_; ++d) s += emb[g * DIN_ + d] * Wout[d * O_ + o];
    out[gid] = s;
}

// ---------------------------------------------------------------------------
extern "C" void kernel_launch(void* const* d_in, const int* in_sizes, int n_in,
                              void* d_out, int out_size, void* d_ws, size_t ws_size,
                              hipStream_t stream) {
    const float* x      = (const float*)d_in[0];
    const int*   ei     = (const int*)d_in[1];
    const int*   batch  = (const int*)d_in[2];
    const float* W1s    = (const float*)d_in[3];
    const float* b1s    = (const float*)d_in[4];
    const float* W2s    = (const float*)d_in[5];
    const float* b2s    = (const float*)d_in[6];
    const float* eps    = (const float*)d_in[7];
    const float* gammas = (const float*)d_in[8];
    const float* betas  = (const float*)d_in[9];
    const float* Wout   = (const float*)d_in[10];
    const float* bout   = (const float*)d_in[11];
    float* out = (float*)d_out;

    char* ws = (char*)d_ws;
    size_t off = 0;
    auto alloc = [&](size_t bytes) {
        void* p = ws + off;
        off = (off + bytes + 255) & ~(size_t)255;
        return p;
    };
    float*  aggz = (float*)alloc((size_t)N_ * H_ * 4);  // agg, aliased as z2 out
    float*  hbuf = (float*)alloc((size_t)N_ * H_ * 4);
    float*  emb  = (float*)alloc((size_t)G_ * DIN_ * 4);
    __bf16* Wp   = (__bf16*)alloc((size_t)2 * L_ * 32 * 512 * 2);
    float*  gsum = (float*)alloc(H_ * 4);
    float*  gsq  = (float*)alloc(H_ * 4);

    hipMemsetAsync(emb, 0, (size_t)G_ * DIN_ * 4, stream);
    pack_weights<<<(2 * L_ * 32 * 32 * 16 + 255) / 256, 256, 0, stream>>>(W1s, W2s, Wp);
    pool_x<<<(N_ * 32 + 255) / 256, 256, 0, stream>>>(x, batch, emb);

    for (int l = 0; l < L_; ++l) {
        const float* hin = (l == 0) ? x : hbuf;
        hipMemsetAsync(aggz, 0, (size_t)N_ * H_ * 4, stream);
        edge_agg<<<(E_ * 32 + 255) / 256, 256, 0, stream>>>(hin, ei, aggz);
        hipMemsetAsync(gsum, 0, H_ * 4, stream);
        hipMemsetAsync(gsq, 0, H_ * 4, stream);
        gin_mlp<<<(N_ + 127) / 128, 256, 0, stream>>>(hin, aggz, Wp, l, b1s, b2s, eps,
                                                      aggz, gsum, gsq);
        bn_relu_pool<<<(N_ * 32 + 255) / 256, 256, 0, stream>>>(aggz, gsum, gsq, gammas,
                                                                betas, l, batch, hbuf, emb);
    }
    classify<<<(G_ * O_ + 255) / 256, 256, 0, stream>>>(emb, Wout, bout, out);
}
